// MultiHeadAttention_27066883899424
// MI455X (gfx1250) — matrix operand order repaired and checked by reading, not compiled, not run on hardware
//
#include <hip/hip_runtime.h>
#include <hip/hip_bf16.h>

typedef __bf16 bf16_t;
typedef __attribute__((ext_vector_type(16))) __bf16 v16bf;
typedef __attribute__((ext_vector_type(8)))  __bf16 v8bf;
typedef __attribute__((ext_vector_type(8)))  float  v8f;
typedef __attribute__((ext_vector_type(4)))  float  f4;
typedef __attribute__((ext_vector_type(4)))  unsigned int u32x4;
typedef __attribute__((ext_vector_type(8)))  int i32x8;
typedef __attribute__((ext_vector_type(4)))  int i32x4;

#if defined(__has_builtin)
#if __has_builtin(__builtin_amdgcn_tensor_load_to_lds) && \
    __has_builtin(__builtin_amdgcn_s_wait_tensorcnt)
#define HAVE_TDM 1
#endif
#endif
#ifndef HAVE_TDM
#define HAVE_TDM 0
#endif

constexpr int B_ = 4, N_ = 2048, M_ = 2048, D_ = 512, H_ = 8, DH_ = 64;
constexpr float SCALE = 0.125f; // 1/sqrt(DH)
// Staged W tile row stride: 512 dwords + TDM pad of 4 dwords per 256 dwords.
constexpr int WPAD = 520;

static __device__ __forceinline__ v16bf cat8(v8bf lo, v8bf hi) {
  return __builtin_shufflevector(lo, hi, 0, 1, 2, 3, 4, 5, 6, 7,
                                 8, 9, 10, 11, 12, 13, 14, 15);
}

// ---------------------------------------------------------------------------
// Projection GEMM:  Out = (Xa (+ Xb)) @ W^T + bias,  bf16 output.
//   TRANS=false: Out[row][col] (row stride D_)  -> Q, K
//   TRANS=true : Out[b][col][m] per-batch       -> V^T for PV B-frags
// All 8 waves of a block share tileN, i.e. the same 16 W rows: stage that
// 16x512 f32 tile into LDS once per block with the Tensor Data Mover
// (tensor_load_to_lds + s_wait_tensorcnt), using TDM padding (4 dw / 256 dw)
// so the 16 lanes' ds_load_b128 B-fragment reads hit distinct bank groups.
// ---------------------------------------------------------------------------
template <bool HAS_XB, bool TRANS>
__global__ __launch_bounds__(256) void proj_gemm_kernel(
    const float* __restrict__ Xa, const float* __restrict__ Xb,
    const float* __restrict__ W, const float* __restrict__ bias,
    bf16_t* __restrict__ Out)
{
  __shared__ float s_w[16 * WPAD];   // 33.3 KB staged W tile

  const int wave = threadIdx.x >> 5;
  const int lane = threadIdx.x & 31;
  const int half = lane >> 4;     // 0: lanes 0-15, 1: lanes 16-31
  const int lr   = lane & 15;

  const int tileM = blockIdx.x * 8 + wave;   // 64 blocks * 8 waves = 512 row tiles
  const int tileN = blockIdx.y;              // 32 col tiles (shared by the block)

#if HAVE_TDM
  if (wave == 0) {
    const unsigned lds_base = (unsigned)(unsigned long long)(size_t)(&s_w[0]);
    const unsigned long long ga =
        (unsigned long long)(size_t)(W + (size_t)(tileN * 16) * D_);
    // D# group 0: count=1 | lds_addr | global_addr[56:0] | type=2
    u32x4 g0 = { 1u, lds_base, (unsigned)ga,
                 (unsigned)((ga >> 32) & 0x1ffffffull) | 0x80000000u };
    // D# group 1: data_size=4B(2), pad_enable, pad_interval=256dw(7),
    //             pad_amount=4dw(3); dims: tensor 512x16, tile 512x16, stride 512.
    i32x8 g1 = { (int)((2u << 16) | (1u << 20) | (7u << 22) | (3u << 25)),
                 (int)(512u << 16),   // tensor_dim0 lo16 @ [31:16]
                 (int)(16u  << 16),   // dim0 hi16 = 0, tensor_dim1 lo16 @ [31:16]
                 (int)(512u << 16),   // dim1 hi16 = 0, tile_dim0 @ [31:16]
                 16,                  // tile_dim1 = 16, tile_dim2 = 0
                 512,                 // tensor_dim0_stride lo32
                 0, 0 };
    i32x4 gz = {0, 0, 0, 0};
#if __clang_major__ >= 23
    i32x8 gz8 = {0, 0, 0, 0, 0, 0, 0, 0};
    __builtin_amdgcn_tensor_load_to_lds(g0, g1, gz, gz, gz8, 0);
#else
    __builtin_amdgcn_tensor_load_to_lds(g0, g1, gz, gz, 0);
#endif
    __builtin_amdgcn_s_wait_tensorcnt(0);
  }
#else
  for (int g = threadIdx.x; g < 16 * 128; g += 256) {   // cooperative fallback
    const int r = g >> 7, kg = (g & 127) * 4;
    *(f4*)&s_w[r * WPAD + kg + ((kg >> 8) << 2)] =
        *(const f4*)(W + ((size_t)(tileN * 16 + r)) * D_ + kg);
  }
#endif
  __syncthreads();

  const int rowA = tileM * 16 + lr;          // A-matrix: lane -> M row
  const float* xa = Xa + (size_t)rowA * D_;
  const float* xb = HAS_XB ? (Xb + (size_t)rowA * D_) : nullptr;
  const float* wrow = &s_w[lr * WPAD];       // B-matrix: lane -> N col (= W row)

  v8f acc = {};
  for (int k0 = 0; k0 < D_; k0 += 32) {
    // A frag (16x32 bf16): lanes 0-15 hold K = k0+{0..15},
    //                      lanes 16-31 hold K = k0+{16..31}
    const int ca0 = k0 + half * 16;
    v8f x0 = *(const v8f*)(xa + ca0);
    v8f x1 = *(const v8f*)(xa + ca0 + 8);
    if constexpr (HAS_XB) {
      x0 += *(const v8f*)(xb + ca0);
      x1 += *(const v8f*)(xb + ca0 + 8);
    }
    v16bf a;
#pragma unroll
    for (int j = 0; j < 8; ++j) { a[j] = (bf16_t)x0[j]; a[j + 8] = (bf16_t)x1[j]; }

    // B frag (32x16 bf16) from the padded LDS tile.
    const int cb0 = k0 + half * 16;
    const int ci  = cb0 + ((cb0 >> 8) << 2);   // account for TDM padding
    const f4 wa = *(const f4*)(wrow + ci);
    const f4 wb = *(const f4*)(wrow + ci + 4);
    const f4 wc = *(const f4*)(wrow + ci + 8);
    const f4 wd = *(const f4*)(wrow + ci + 12);
    v16bf bfr;
#pragma unroll
    for (int j = 0; j < 4; ++j) {
      bfr[j]      = (bf16_t)wa[j];
      bfr[j + 4]  = (bf16_t)wb[j];
      bfr[j + 8]  = (bf16_t)wc[j];
      bfr[j + 12] = (bf16_t)wd[j];
    }

    acc = __builtin_amdgcn_wmma_f32_16x16x32_bf16(false, a, false, bfr,
                                                  (short)0, acc, false, false);
  }

  const float bv = bias[tileN * 16 + lr];
#pragma unroll
  for (int j = 0; j < 8; ++j) {
    const int r = tileM * 16 + j + 8 * half;  // C layout: VGPR j -> row j+8*half
    const int c = tileN * 16 + lr;            // lane%16 -> col
    const float val = acc[j] + bv;
    if constexpr (!TRANS) {
      Out[(size_t)r * D_ + c] = (bf16_t)val;
    } else {
      const int bb = r >> 11;                 // r / 2048
      const int m  = r & 2047;
      Out[((size_t)bb * D_ + c) * M_ + m] = (bf16_t)val;
    }
  }
}

// ---------------------------------------------------------------------------
// Attention: one WG per (b, h, 16 query rows). 8 waves / 256 threads.
//   Phase 1: scores (16 x 2048) via WMMA into LDS, fused scale/factor/mask
//            (unconditional loads + v_cndmask select -> no exec branching).
//   Phase 2: softmax over M in LDS (float4 sweeps), one coalesced attn write.
//   Phase 3: attn @ V via WMMA, 8-wave LDS reduction -> hidden.
// ---------------------------------------------------------------------------
__global__ __launch_bounds__(256) void attn_kernel(
    const bf16_t* __restrict__ Q, const bf16_t* __restrict__ K,
    const bf16_t* __restrict__ VT,
    const float* __restrict__ key_weights,
    const unsigned char* __restrict__ key_masks,
    const float* __restrict__ attn_factors,
    const unsigned char* __restrict__ attn_masks,
    float* __restrict__ hidden, float* __restrict__ attn_out)
{
  extern __shared__ char smem_raw[];
  float* s_scores = (float*)smem_raw;            // [16][M_]  = 128 KB
  float* s_red    = s_scores + 16 * M_;          // [256]
  float* s_rowmax = s_red + 256;                 // [16]
  float* s_rowsum = s_rowmax + 16;               // [16]
  float* s_pv     = s_rowsum + 16;               // [8][16][64] = 32 KB

  const int b = blockIdx.z, h = blockIdx.y;
  const int n0 = blockIdx.x * 16;
  const int t = threadIdx.x;
  const int wave = t >> 5, lane = t & 31;
  const int half = lane >> 4, lr = lane & 15;

  const bf16_t* qh = Q + ((size_t)(b * N_ + n0)) * D_ + h * DH_;   // row stride D_
  const bf16_t* kh = K + (size_t)b * M_ * D_ + h * DH_;            // row stride D_
  const bf16_t* vt = VT + (size_t)b * D_ * M_ + (size_t)(h * DH_) * M_; // [d][m]

  // ---------------- Phase 1: scores into LDS ----------------
  v16bf aq[2];
#pragma unroll
  for (int s = 0; s < 2; ++s) {
    const bf16_t* qrow = qh + (size_t)lr * D_ + s * 32 + half * 16;
    aq[s] = cat8(*(const v8bf*)(qrow), *(const v8bf*)(qrow + 8));
  }

  for (int i = 0; i < 16; ++i) {
    const int mcol = (wave + 8 * i) * 16 + lr;
    const bf16_t* krow = kh + (size_t)mcol * D_ + half * 16;
    if (i + 1 < 16) {   // prefetch next K tile -> global_prefetch_b8
      const int mnext = (wave + 8 * (i + 1)) * 16 + lr;
      __builtin_prefetch(kh + (size_t)mnext * D_, 0, 1);
    }
    v16bf b0 = cat8(*(const v8bf*)(krow),      *(const v8bf*)(krow + 8));
    v16bf b1 = cat8(*(const v8bf*)(krow + 32), *(const v8bf*)(krow + 40));

    v8f c8 = {};
    c8 = __builtin_amdgcn_wmma_f32_16x16x32_bf16(false, aq[0], false, b0,
                                                 (short)0, c8, false, false);
    c8 = __builtin_amdgcn_wmma_f32_16x16x32_bf16(false, aq[1], false, b1,
                                                 (short)0, c8, false, false);

    // Fused epilogue: all loads unconditional, constant-offset addressed.
    const float    kws = SCALE * key_weights[b * M_ + mcol];
    const unsigned kmv = key_masks[b * M_ + mcol];
    const size_t   rb  = (size_t)(b * N_ + n0 + 8 * half) * M_ + mcol;
    const float*         fp = attn_factors + rb;
    const unsigned char* mp = attn_masks + rb;
    float*               sp = s_scores + (8 * half) * M_ + mcol;
#pragma unroll
    for (int j = 0; j < 8; ++j) {
      const float    fct = fp[(size_t)j * M_];
      const unsigned am  = (unsigned)mp[(size_t)j * M_] | kmv;
      const float    sc  = c8[j] * kws * fct;
      sp[j * M_] = am ? -__builtin_inff() : sc;
    }
  }
  __syncthreads();

  // ---------------- Phase 2: softmax over M (float4 sweeps) ----------------
  const int rrow = t >> 4;   // 16 threads cooperate per row
  const int rl   = t & 15;
  float mx = -__builtin_inff();
  for (int i = 0; i < M_ / 64; ++i) {
    const f4 v = *(const f4*)&s_scores[rrow * M_ + rl * 4 + 64 * i];
    mx = fmaxf(mx, fmaxf(fmaxf(v.x, v.y), fmaxf(v.z, v.w)));
  }
  s_red[t] = mx;
  __syncthreads();
  if (rl == 0) {
    float v = -__builtin_inff();
#pragma unroll
    for (int j = 0; j < 16; ++j) v = fmaxf(v, s_red[rrow * 16 + j]);
    s_rowmax[rrow] = v;
  }
  __syncthreads();
  const float rm = s_rowmax[rrow];
  float sum = 0.f;
  for (int i = 0; i < M_ / 64; ++i) {
    float* p = &s_scores[rrow * M_ + rl * 4 + 64 * i];
    f4 v = *(const f4*)p;
    v.x = __expf(v.x - rm); v.y = __expf(v.y - rm);
    v.z = __expf(v.z - rm); v.w = __expf(v.w - rm);
    *(f4*)p = v;
    sum += v.x + v.y + v.z + v.w;
  }
  s_red[t] = sum;
  __syncthreads();
  if (rl == 0) {
    float v = 0.f;
#pragma unroll
    for (int j = 0; j < 16; ++j) v += s_red[rrow * 16 + j];
    s_rowsum[rrow] = v;
  }
  __syncthreads();

  // Normalize in LDS and write attn: 256 threads x float4 -> b128 stores.
  float* arow = attn_out + ((size_t)(b * H_ + h) * N_ + n0) * M_;
  for (int r = 0; r < 16; ++r) {
    const float inv = 1.f / s_rowsum[r];
#pragma unroll
    for (int i = 0; i < M_ / 1024; ++i) {
      const int m = (i * 256 + t) * 4;
      f4 v = *(const f4*)&s_scores[r * M_ + m];
      v *= inv;
      *(f4*)&s_scores[r * M_ + m] = v;
      *(f4*)&arow[(size_t)r * M_ + m] = v;
    }
  }
  __syncthreads();

  // ---------------- Phase 3: hidden = attn @ V ----------------
  v8f pacc[4] = {v8f{}, v8f{}, v8f{}, v8f{}};
  const int mwbase = wave * 256;
  for (int ks = 0; ks < 256; ks += 32) {
    const int mb = mwbase + ks;
    const v8f s0 = *(const v8f*)&s_scores[lr * M_ + mb + half * 16];
    const v8f s1 = *(const v8f*)&s_scores[lr * M_ + mb + half * 16 + 8];
    v16bf a;
#pragma unroll
    for (int j = 0; j < 8; ++j) { a[j] = (bf16_t)s0[j]; a[j + 8] = (bf16_t)s1[j]; }
#pragma unroll
    for (int dt = 0; dt < 4; ++dt) {
      const bf16_t* vrow = vt + (size_t)(dt * 16 + lr) * M_ + mb + half * 16;
      const v16bf bb = cat8(*(const v8bf*)(vrow), *(const v8bf*)(vrow + 8));
      pacc[dt] = __builtin_amdgcn_wmma_f32_16x16x32_bf16(false, a, false, bb,
                                                         (short)0, pacc[dt],
                                                         false, false);
    }
  }
#pragma unroll
  for (int dt = 0; dt < 4; ++dt) {
#pragma unroll
    for (int j = 0; j < 8; ++j) {
      const int r = j + 8 * half;
      const int d = dt * 16 + lr;
      s_pv[(wave * 16 + r) * 64 + d] = pacc[dt][j];
    }
  }
  __syncthreads();

  // Reduce 8 wave-partials; 16 rows x 16 threads x float4 -> b128 stores.
  {
    const int r = t >> 4;
    const int d0 = (t & 15) * 4;
    f4 ssum = {0.f, 0.f, 0.f, 0.f};
#pragma unroll
    for (int w = 0; w < 8; ++w) ssum += *(const f4*)&s_pv[(w * 16 + r) * 64 + d0];
    *(f4*)&hidden[((size_t)(b * N_ + n0 + r)) * D_ + h * DH_ + d0] = ssum;
  }
}

// ---------------------------------------------------------------------------
extern "C" void kernel_launch(void* const* d_in, const int* in_sizes, int n_in,
                              void* d_out, int out_size, void* d_ws, size_t ws_size,
                              hipStream_t stream) {
  const float* input_q = (const float*)d_in[0];
  const float* input_k = (const float*)d_in[1];
  const float* input_v = (const float*)d_in[2];
  const float* pos_q   = (const float*)d_in[3];
  const float* pos_k   = (const float*)d_in[4];
  const float* key_w   = (const float*)d_in[5];
  const unsigned char* key_m = (const unsigned char*)d_in[6];
  const float* att_f   = (const float*)d_in[7];
  const unsigned char* att_m = (const unsigned char*)d_in[8];
  const float* Wq = (const float*)d_in[9];
  const float* bq = (const float*)d_in[10];
  const float* Wk = (const float*)d_in[11];
  const float* bk = (const float*)d_in[12];
  const float* Wv = (const float*)d_in[13];
  const float* bv = (const float*)d_in[14];

  // Workspace: bf16 Q, K (row-major) and V^T (per-batch [d][m]) = 24 MB total.
  bf16_t* qws  = (bf16_t*)d_ws;
  bf16_t* kws  = qws + (size_t)B_ * N_ * D_;
  bf16_t* vtws = kws + (size_t)B_ * M_ * D_;

  dim3 pgrid(64, 32);   // 512 row tiles (8 waves/block) x 32 col tiles
  proj_gemm_kernel<true,  false><<<pgrid, 256, 0, stream>>>(input_q, pos_q, Wq, bq, qws);
  proj_gemm_kernel<true,  false><<<pgrid, 256, 0, stream>>>(input_k, pos_k, Wk, bk, kws);
  proj_gemm_kernel<false, true ><<<pgrid, 256, 0, stream>>>(input_v, nullptr, Wv, bv, vtws);

  float* hidden = (float*)d_out;
  float* attn   = hidden + (size_t)B_ * N_ * D_;

  const size_t smem =
      (size_t)(16 * M_ + 256 + 16 + 16 + 8 * 16 * 64) * sizeof(float); // ~161 KB
  dim3 agrid(N_ / 16, H_, B_);
  attn_kernel<<<agrid, 256, smem, stream>>>(qws, kws, vtws, key_w, key_m,
                                            att_f, att_m, hidden, attn);
}
